// MaskedAttentionHead_9783935500852
// MI455X (gfx1250) — matrix operand order repaired and checked by reading, not compiled, not run on hardware
//
#include <hip/hip_runtime.h>
#include <hip/hip_bf16.h>

// ---------------------------------------------------------------------------
// MaskedAttentionHead for MI455X (gfx1250, wave32, WMMA + TDM).
// Pipeline: f32->bf16 casts; QKV projections (bf16 WMMA GEMM, V stored
// transposed); S = QK^T/32 (masked tile-skip); row softmax (wave per row);
// out = P @ V (causal K-limit). All GEMMs share one A*B^T WMMA kernel whose
// LDS tiles are double-buffered and filled by the Tensor Data Mover
// (tensor_load_to_lds, TENSORcnt), overlapping DMA of tile i+1 with WMMA
// compute of tile i.
// ---------------------------------------------------------------------------

typedef __bf16 bf16_t;
typedef bf16_t v4bf  __attribute__((ext_vector_type(4)));
typedef bf16_t v8bf  __attribute__((ext_vector_type(8)));
typedef bf16_t v16bf __attribute__((ext_vector_type(16)));
typedef float  v8f   __attribute__((ext_vector_type(8)));
typedef unsigned int v4u  __attribute__((ext_vector_type(4)));
typedef int          v4i_t __attribute__((ext_vector_type(4)));
typedef int          v8i_t __attribute__((ext_vector_type(8)));

#define BM 128
#define BN 128
#define BK 32
#define D_MODEL 1024
#define SEQ 2048
#define NBATCH 4

enum { EPI_BF16 = 0, EPI_VT = 1, EPI_SCORES = 2, EPI_F32 = 3 };

union FragU { v8bf h[2]; v16bf v; };
union PackU { v8bf v; float4 f; };

// ---------------------------------------------------------------------------
// f32 -> bf16 cast, 4 elements / thread (global_load_b128 + 8B store)
// ---------------------------------------------------------------------------
__global__ __launch_bounds__(256) void f32_to_bf16_kernel(
    const float* __restrict__ src, bf16_t* __restrict__ dst, int n4) {
  int i = blockIdx.x * 256 + threadIdx.x;
  if (i < n4) {
    float4 vv = ((const float4*)src)[i];
    v4bf o;
    o[0] = (bf16_t)vv.x; o[1] = (bf16_t)vv.y;
    o[2] = (bf16_t)vv.z; o[3] = (bf16_t)vv.w;
    ((v4bf*)dst)[i] = o;
  }
}

// ---------------------------------------------------------------------------
// TDM: DMA one [BM rows x BK elements] bf16 tile (row-major, stride_elts
// between rows) from global memory into LDS at lds_addr. D# layout per
// CDNA5 ISA ch.8: group0 = count/lds_addr/global_addr/type, group1 =
// data_size + tensor dims + tile dims + dim0 stride. All operands are
// wave-uniform; EXEC is ignored for tensor instructions.
// ---------------------------------------------------------------------------
__device__ __forceinline__ void tdm_load_tile_2d(unsigned lds_addr,
                                                 const bf16_t* gptr,
                                                 unsigned tensor_rows,
                                                 unsigned stride_elts) {
  unsigned long long ga = (unsigned long long)(uintptr_t)gptr;
  v4u g0;
  g0[0] = 1u;                                         // count=1, user mode
  g0[1] = lds_addr;                                   // lds_addr[31:0]
  g0[2] = (unsigned)(ga & 0xFFFFFFFFull);             // global_addr[31:0]
  g0[3] = (unsigned)((ga >> 32) & 0x01FFFFFFull)      // global_addr[56:32]
          | 0x80000000u;                              // type=2 ("image")
  v8i_t g1;
  g1[0] = 0x00010000;                                 // data_size=1 -> 2 bytes
  g1[1] = (int)((stride_elts & 0xFFFFu) << 16);       // tensor_dim0 lo16
  g1[2] = (int)(((stride_elts >> 16) & 0xFFFFu) |     // tensor_dim0 hi16
                ((tensor_rows & 0xFFFFu) << 16));     // tensor_dim1 lo16
  g1[3] = (int)(((tensor_rows >> 16) & 0xFFFFu) |     // tensor_dim1 hi16
                ((unsigned)BK << 16));                // tile_dim0 = 32
  g1[4] = BM;                                         // tile_dim1=128, dim2=0
  g1[5] = (int)stride_elts;                           // tensor_dim0_stride lo32
  g1[6] = 0;                                          // stride hi16, dim1_stride
  g1[7] = 0;
  v4i_t z4 = (v4i_t)0;
#if defined(__clang_major__) && (__clang_major__ >= 23)
  v8i_t z8 = (v8i_t)0;
  __builtin_amdgcn_tensor_load_to_lds(g0, g1, z4, z4, z8, 0);
#else
  __builtin_amdgcn_tensor_load_to_lds(g0, g1, z4, z4, 0);
#endif
}

// ---------------------------------------------------------------------------
// Tiled GEMM  C = A * B^T   (A: [M x K] row-major, B: [N x K] row-major,
// contraction contiguous for both -> every matmul in this problem fits).
// 256 threads = 8 wave32s; each wave owns a 32x64 slab = 2x4 WMMA tiles.
// LDS tiles double-buffered; TDM fills buffer cur^1 while waves compute on
// buffer cur. Fragment gathers follow the CDNA5 16-bit A/B VGPR layouts.
// ---------------------------------------------------------------------------
template <int EPI>
__global__ __launch_bounds__(256) void gemm_abT_kernel(
    const bf16_t* __restrict__ A, const bf16_t* __restrict__ B,
    void* __restrict__ Cvoid, int Kdim, int lda, int ldb, int ldc, float scale,
    long long strideA, long long strideB, long long strideC) {
  const int tc = blockIdx.x;   // column tile
  const int tr = blockIdx.y;   // row tile
  const int bz = blockIdx.z;   // batch

  // scores: tiles fully above the (j <= i+1) band contribute nothing
  if (EPI == EPI_SCORES && tc > tr + 1) return;

  const bf16_t* Ab = A + (size_t)bz * (size_t)strideA;
  const bf16_t* Bb = B + (size_t)bz * (size_t)strideB;

  const int rowBase = tr * BM;
  const int colBase = tc * BN;

  int kEnd = Kdim;
  if (EPI == EPI_F32) {  // out = P @ V: P columns beyond rowBase+128 are zero
    int lim = ((rowBase + BM + 1 + (BK - 1)) / BK) * BK;
    kEnd = lim < Kdim ? lim : Kdim;
  }
  const int nIter = kEnd / BK;

  __shared__ bf16_t As[2][BM * BK];   // [buf][m][k], 2 x 8 KB
  __shared__ bf16_t Bs[2][BN * BK];   // [buf][n][k], 2 x 8 KB

  const int tid   = threadIdx.x;
  const int lane  = tid & 31;
  const int wid   = tid >> 5;
  const int waveM = (wid >> 1) * 32;   // 4 row slabs
  const int waveN = (wid & 1) * 64;    // 2 col slabs

  // Tile bases in global memory (wave-uniform, advance by BK per iteration).
  const bf16_t* gA = Ab + (size_t)rowBase * (size_t)lda;
  const bf16_t* gB = Bb + (size_t)colBase * (size_t)ldb;

  const unsigned ldsA0 = (unsigned)(uintptr_t)&As[0][0];
  const unsigned ldsA1 = (unsigned)(uintptr_t)&As[1][0];
  const unsigned ldsB0 = (unsigned)(uintptr_t)&Bs[0][0];
  const unsigned ldsB1 = (unsigned)(uintptr_t)&Bs[1][0];

  v8f acc[2][4];
#pragma unroll
  for (int fm = 0; fm < 2; ++fm)
#pragma unroll
    for (int fn = 0; fn < 4; ++fn) acc[fm][fn] = (v8f)0.0f;

  // Prologue: DMA first tile pair into buffer 0 (wave 0 issues; per-wave
  // TENSORcnt, so wave 0 also waits; barriers release the others).
  if (tid < 32) {
    tdm_load_tile_2d(ldsA0, gA, BM, (unsigned)lda);
    tdm_load_tile_2d(ldsB0, gB, BN, (unsigned)ldb);
  }

  for (int it = 0; it < nIter; ++it) {
    const int cur = it & 1;
    const int k0  = it * BK;

    __syncthreads();  // all waves done reading buffer cur^1 from prev iter
    if (tid < 32) {
      if (it + 1 < nIter) {
        // DMA next tile pair into the other buffer, then wait until only
        // those 2 ops remain outstanding => current buffer is complete.
        tdm_load_tile_2d(cur ? ldsA0 : ldsA1, gA + k0 + BK, BM, (unsigned)lda);
        tdm_load_tile_2d(cur ? ldsB0 : ldsB1, gB + k0 + BK, BN, (unsigned)ldb);
        __builtin_amdgcn_s_wait_tensorcnt(2);
      } else {
        __builtin_amdgcn_s_wait_tensorcnt(0);
      }
    }
    __syncthreads();  // buffer cur is ready for all waves

    const bf16_t* Asb = As[cur];
    const bf16_t* Bsb = Bs[cur];

    // A frag: lane L holds row M = L&15; K = (L>>4)*8 + {0..7, 16..23}
    FragU af[2];
#pragma unroll
    for (int fm = 0; fm < 2; ++fm) {
      int m  = waveM + fm * 16 + (lane & 15);
      int kh = (lane >> 4) * 8;
      af[fm].h[0] = *(const v8bf*)(Asb + m * BK + kh);
      af[fm].h[1] = *(const v8bf*)(Asb + m * BK + kh + 16);
    }
    // B frag: lane L holds col N = L&15; K = (L>>4)*8 + {0..7, 16..23}
    FragU bf[4];
#pragma unroll
    for (int fn = 0; fn < 4; ++fn) {
      int n  = waveN + fn * 16 + (lane & 15);
      int kb = (lane >> 4) * 8;
      bf[fn].h[0] = *(const v8bf*)(Bsb + n * BK + kb);
      bf[fn].h[1] = *(const v8bf*)(Bsb + n * BK + kb + 16);
    }
#pragma unroll
    for (int fm = 0; fm < 2; ++fm)
#pragma unroll
      for (int fn = 0; fn < 4; ++fn)
        acc[fm][fn] = __builtin_amdgcn_wmma_f32_16x16x32_bf16(
            false, af[fm].v, false, bf[fn].v, (short)0, acc[fm][fn],
            false, false);
  }

  // D layout: lane L, VGPR r -> row M = (L>>4)*8 + r, col N = L&15
  const int hi = lane >> 4;
  const int ln = lane & 15;

  if (EPI == EPI_BF16) {
    bf16_t* C = (bf16_t*)Cvoid;
#pragma unroll
    for (int fm = 0; fm < 2; ++fm)
#pragma unroll
      for (int fn = 0; fn < 4; ++fn)
#pragma unroll
        for (int r = 0; r < 8; ++r) {
          int gr = rowBase + waveM + fm * 16 + hi * 8 + r;
          int gc = colBase + waveN + fn * 16 + ln;
          C[(size_t)gr * ldc + gc] = (bf16_t)acc[fm][fn][r];
        }
  } else if (EPI == EPI_VT) {
    // V^T store: Vt[b][e][t], per-lane 8 contiguous t -> one 16B store
    bf16_t* C = (bf16_t*)Cvoid;
#pragma unroll
    for (int fm = 0; fm < 2; ++fm)
#pragma unroll
      for (int fn = 0; fn < 4; ++fn) {
        int gr0 = rowBase + waveM + fm * 16 + hi * 8;  // global token row
        int gc  = colBase + waveN + fn * 16 + ln;      // feature e
        int b   = gr0 >> 11;                           // / SEQ
        int tt  = gr0 & (SEQ - 1);
        PackU p;
#pragma unroll
        for (int r = 0; r < 8; ++r) p.v[r] = (bf16_t)acc[fm][fn][r];
        *(float4*)(C + (size_t)b * (size_t)strideC + (size_t)gc * ldc + tt) =
            p.f;
      }
  } else if (EPI == EPI_SCORES) {
    bf16_t* C = (bf16_t*)Cvoid + (size_t)bz * (size_t)strideC;
#pragma unroll
    for (int fm = 0; fm < 2; ++fm)
#pragma unroll
      for (int fn = 0; fn < 4; ++fn)
#pragma unroll
        for (int r = 0; r < 8; ++r) {
          int gr = rowBase + waveM + fm * 16 + hi * 8 + r;
          int gc = colBase + waveN + fn * 16 + ln;
          C[(size_t)gr * ldc + gc] = (bf16_t)(acc[fm][fn][r] * scale);
        }
  } else {  // EPI_F32: final output
    float* C = (float*)Cvoid + (size_t)bz * (size_t)strideC;
#pragma unroll
    for (int fm = 0; fm < 2; ++fm)
#pragma unroll
      for (int fn = 0; fn < 4; ++fn)
#pragma unroll
        for (int r = 0; r < 8; ++r) {
          int gr = rowBase + waveM + fm * 16 + hi * 8 + r;
          int gc = colBase + waveN + fn * 16 + ln;
          C[(size_t)gr * ldc + gc] = acc[fm][fn][r];
        }
  }
}

// ---------------------------------------------------------------------------
// Row softmax with j <= i+1 mask; one wave32 per row; P written in place
// (masked entries zeroed so the PV GEMM can read dense rows).
// ---------------------------------------------------------------------------
__global__ __launch_bounds__(256) void softmax_rows_kernel(bf16_t* __restrict__ S,
                                                           int T) {
  const int wid  = threadIdx.x >> 5;
  const int lane = threadIdx.x & 31;
  const int row  = blockIdx.x * 8 + wid;           // 0 .. B*T-1
  const int i    = row & (SEQ - 1);                // query index in batch
  const int jmax = i + 1;                          // inclusive mask bound
  bf16_t* rp = S + (size_t)row * T;

  float m = -3.0e38f;
  for (int j = lane; j < T; j += 32)
    if (j <= jmax) m = fmaxf(m, (float)rp[j]);
#pragma unroll
  for (int off = 16; off > 0; off >>= 1) m = fmaxf(m, __shfl_xor(m, off, 32));

  float l = 0.0f;
  for (int j = lane; j < T; j += 32)
    if (j <= jmax) l += __expf((float)rp[j] - m);
#pragma unroll
  for (int off = 16; off > 0; off >>= 1) l += __shfl_xor(l, off, 32);
  const float inv = 1.0f / l;

  for (int j = lane; j < T; j += 32) {
    float p = (j <= jmax) ? __expf((float)rp[j] - m) * inv : 0.0f;
    rp[j] = (bf16_t)p;
  }
}

// ---------------------------------------------------------------------------
// Host-side orchestration (all launches on `stream`; graph-capture safe).
// Workspace layout (bf16): xb 16M | Wq/Wk/Wv 2M each | Q 16M | K 16M |
// Vt 16M | S/P 32M  => ~102 MiB.
// ---------------------------------------------------------------------------
extern "C" void kernel_launch(void* const* d_in, const int* in_sizes, int n_in,
                              void* d_out, int out_size, void* d_ws,
                              size_t ws_size, hipStream_t stream) {
  (void)in_sizes; (void)n_in; (void)out_size; (void)ws_size;
  const float* x  = (const float*)d_in[0];
  const float* Wq = (const float*)d_in[1];
  const float* Wk = (const float*)d_in[2];
  const float* Wv = (const float*)d_in[3];
  float* out = (float*)d_out;

  const int Bn = NBATCH, T = SEQ, d = D_MODEL;
  const size_t Nrow = (size_t)Bn * T;

  char* w = (char*)d_ws;
  bf16_t* xb  = (bf16_t*)w; w += Nrow * d * sizeof(bf16_t);
  bf16_t* wqb = (bf16_t*)w; w += (size_t)d * d * sizeof(bf16_t);
  bf16_t* wkb = (bf16_t*)w; w += (size_t)d * d * sizeof(bf16_t);
  bf16_t* wvb = (bf16_t*)w; w += (size_t)d * d * sizeof(bf16_t);
  bf16_t* Qb  = (bf16_t*)w; w += Nrow * d * sizeof(bf16_t);
  bf16_t* Kb  = (bf16_t*)w; w += Nrow * d * sizeof(bf16_t);
  bf16_t* Vt  = (bf16_t*)w; w += Nrow * d * sizeof(bf16_t);   // [b][e][t]
  bf16_t* S   = (bf16_t*)w;                                   // [b][q][k]

  // 1) casts
  {
    int n4 = (int)(Nrow * d / 4);
    f32_to_bf16_kernel<<<(n4 + 255) / 256, 256, 0, stream>>>(x, xb, n4);
    int w4 = d * d / 4;
    f32_to_bf16_kernel<<<(w4 + 255) / 256, 256, 0, stream>>>(Wq, wqb, w4);
    f32_to_bf16_kernel<<<(w4 + 255) / 256, 256, 0, stream>>>(Wk, wkb, w4);
    f32_to_bf16_kernel<<<(w4 + 255) / 256, 256, 0, stream>>>(Wv, wvb, w4);
  }

  // 2) projections: Y[t,e] = sum_d x[t,d] * W[e,d]  (A*B^T, weights shared)
  dim3 gproj(d / BN, (unsigned)(Nrow / BM), 1);
  gemm_abT_kernel<EPI_BF16><<<gproj, 256, 0, stream>>>(
      xb, wqb, Qb, d, d, d, d, 1.0f, 0, 0, 0);
  gemm_abT_kernel<EPI_BF16><<<gproj, 256, 0, stream>>>(
      xb, wkb, Kb, d, d, d, d, 1.0f, 0, 0, 0);
  gemm_abT_kernel<EPI_VT><<<gproj, 256, 0, stream>>>(
      xb, wvb, Vt, d, d, d, /*ldc=*/T, 1.0f, 0, 0,
      /*batch strideC=*/(long long)d * T);

  // 3) scores: S[b][q][k] = Q[b,q,:] . K[b,k,:] / 32, band tiles only
  dim3 gs(T / BN, T / BM, Bn);
  gemm_abT_kernel<EPI_SCORES><<<gs, 256, 0, stream>>>(
      Qb, Kb, S, d, d, d, T, 1.0f / 32.0f, (long long)T * d,
      (long long)T * d, (long long)T * T);

  // 4) masked softmax over rows, P in place
  softmax_rows_kernel<<<(unsigned)(Nrow / 8), 256, 0, stream>>>(S, T);

  // 5) out[b] = P[b] @ V[b]  (B operand is Vt, causal K-limit in kernel)
  dim3 go(d / BN, T / BM, Bn);
  gemm_abT_kernel<EPI_F32><<<go, 256, 0, stream>>>(
      S, Vt, out, T, T, T, d, 1.0f, (long long)T * T, (long long)d * T,
      (long long)T * d);
}
